// BiMPM_6373731467677
// MI455X (gfx1250) — matrix operand order repaired and checked by reading, not compiled, not run on hardware
//
#include <hip/hip_runtime.h>

// ---------------- problem dimensions ----------------
#define Bq    256
#define Sq    64
#define Eemb  300
#define EP    320      // E padded to multiple of 32 (WMMA K-chunk)
#define Hh    100
#define G4    400      // 4*H gate width
#define NPAD  448      // G4 padded to multiple of 64 (GEMM N tile)
#define Kp    20       // perspectives
#define F8K   160      // 8*K match features
#define Mrows (Bq*Sq)  // 16384

typedef __attribute__((ext_vector_type(16))) __bf16 v16bf;
typedef __attribute__((ext_vector_type(4)))  __bf16 bf16x4;
typedef __attribute__((ext_vector_type(8)))  float  f32x8;

// ---------------- WMMA helpers (gfx1250, wave32) ----------------
__device__ __forceinline__ f32x8 zero8() {
  f32x8 z;
#pragma unroll
  for (int i = 0; i < 8; ++i) z[i] = 0.f;
  return z;
}

// A-fragment (16x32 bf16, MxK) from an LDS tile (contiguous K order, matches B):
// lanes 0-15 row M=lane, K in {k0..k0+15};
// lanes 16-31 row M=lane-16, K in {k0+16..k0+31}.
__device__ __forceinline__ v16bf load_a_frag(const __bf16* base, int ld, int m0, int k0, int lane) {
  int row = m0 + (lane & 15);
  int kb  = k0 + ((lane >> 4) << 4);
  const __bf16* p = base + row * ld + kb;
  v16bf a;
#pragma unroll
  for (int e = 0; e < 8; ++e) a[e] = p[e];
#pragma unroll
  for (int e = 0; e < 8; ++e) a[8 + e] = p[8 + e];
  return a;
}

// A-fragment straight from padded f32 global memory (row stride ldk, 16B aligned).
__device__ __forceinline__ v16bf load_a_frag_f32(const float* A, int ldk, int row, int k0, int lane) {
  const float* p = A + (long)row * ldk + k0 + ((lane >> 4) << 4);
  float4 x0 = *(const float4*)(p);
  float4 x1 = *(const float4*)(p + 4);
  float4 x2 = *(const float4*)(p + 8);
  float4 x3 = *(const float4*)(p + 12);
  v16bf a;
  a[0] = (__bf16)x0.x;  a[1] = (__bf16)x0.y;  a[2]  = (__bf16)x0.z;  a[3]  = (__bf16)x0.w;
  a[4] = (__bf16)x1.x;  a[5] = (__bf16)x1.y;  a[6]  = (__bf16)x1.z;  a[7]  = (__bf16)x1.w;
  a[8] = (__bf16)x2.x;  a[9] = (__bf16)x2.y;  a[10] = (__bf16)x2.z;  a[11] = (__bf16)x2.w;
  a[12] = (__bf16)x3.x; a[13] = (__bf16)x3.y; a[14] = (__bf16)x3.z;  a[15] = (__bf16)x3.w;
  return a;
}

// B-fragment (32x16 bf16, KxN) from an [N][K]-major bf16 array.
// Per-lane data is 16 contiguous bf16 = one aligned 32B vector load.
__device__ __forceinline__ v16bf load_b_frag_nk(const __bf16* base, int ld, int n0, int k0, int lane) {
  const __bf16* p = base + (long)(n0 + (lane & 15)) * ld + k0 + ((lane >> 4) << 4);
  return *(const v16bf*)p;
}

__device__ __forceinline__ f32x8 wmma_bf16(v16bf a, v16bf b, f32x8 c) {
  // (neg_a, A, neg_b, B, c_mod, C, reuse_a, reuse_b)
  return __builtin_amdgcn_wmma_f32_16x16x32_bf16(false, a, false, b, (short)0, c, false, false);
}

__device__ __forceinline__ float sigf(float x) { return 1.f / (1.f + expf(-x)); }

// ---------------- embedding gather (writes K-padded rows) ----------------
__global__ void k_embed(const int* __restrict__ q, const float* __restrict__ emb,
                        float* __restrict__ out) {
  int bs = blockIdx.x;
  long row = q[bs];
  for (int v = threadIdx.x; v < EP / 4; v += blockDim.x) {
    int h = v * 4;
    float4 val = (h < Eemb) ? *(const float4*)&emb[row * Eemb + h]
                            : make_float4(0.f, 0.f, 0.f, 0.f);
    *(float4*)&out[(long)bs * EP + h] = val;
  }
}

// ---------------- _reverse_valid (F multiple of 4) ----------------
__global__ void k_reverse(const float* __restrict__ in, float* __restrict__ out,
                          const int* __restrict__ lens, int F) {
  int bs = blockIdx.x;
  int b = bs >> 6, s = bs & 63;
  int idx = lens[b] - 1 - s;
  for (int v = threadIdx.x; v < F / 4; v += blockDim.x) {
    float4 val = (idx >= 0) ? *(const float4*)&in[((long)b * Sq + idx) * F + v * 4]
                            : make_float4(0.f, 0.f, 0.f, 0.f);
    *(float4*)&out[(long)bs * F + v * 4] = val;
  }
}

// zero padded timesteps in place (fw output masking)
__global__ void k_mask(float* __restrict__ x, const int* __restrict__ lens, int F) {
  int bs = blockIdx.x;
  int b = bs >> 6, s = bs & 63;
  if (s < lens[b]) return;
  float4 z = make_float4(0.f, 0.f, 0.f, 0.f);
  for (int v = threadIdx.x; v < F / 4; v += blockDim.x)
    *(float4*)&x[(long)bs * F + v * 4] = z;
}

// ---------------- weight pre-pack: f32 [N][K] -> bf16 [NPAD][Kpad] ----------------
__global__ void k_pack_w(const float* __restrict__ W, __bf16* __restrict__ dst,
                         int N, int K, int Kpad) {
  for (int idx = blockIdx.x * blockDim.x + threadIdx.x; idx < NPAD * Kpad;
       idx += gridDim.x * blockDim.x) {
    int n = idx / Kpad, k = idx % Kpad;
    dst[idx] = (n < N && k < K) ? (__bf16)W[(long)n * K + k] : (__bf16)0.f;
  }
}

// Whh -> bf16 [400][128] (K padded)
__global__ void k_pack_whh(const float* __restrict__ Whh, __bf16* __restrict__ dst) {
  for (int idx = blockIdx.x * blockDim.x + threadIdx.x; idx < G4 * 128;
       idx += gridDim.x * blockDim.x) {
    int n = idx >> 7, k = idx & 127;
    dst[idx] = (k < Hh) ? (__bf16)Whh[n * Hh + k] : (__bf16)0.f;
  }
}

// ---------------- WMMA GEMM: C[M,400] = A[M,Kpad] * Wp[448,Kpad]^T + bias ----------------
// A: padded f32, no tails. Wp: pre-packed bf16 (L2-resident), loaded directly
// into B-fragments — no LDS, no barriers, pure load->cvt->wmma stream.
// 256 threads = 8 waves; 64x64 tile; wave grid 4x2, each wave -> 16x32 of C.
__global__ void k_gemm_bias(const float* __restrict__ A, const __bf16* __restrict__ Wp,
                            const float* __restrict__ bias, float* __restrict__ C,
                            int Kpad) {
  int m0 = blockIdx.x * 64, n0 = blockIdx.y * 64;
  int tid = threadIdx.x, wave = tid >> 5, lane = tid & 31;
  int wm = (wave & 3) * 16;
  int wn = (wave >> 2) * 32;
  int rowA = m0 + wm + (lane & 15);
  int colB = n0 + wn;
  f32x8 acc0 = zero8(), acc1 = zero8();
  for (int k0 = 0; k0 < Kpad; k0 += 32) {
    v16bf a  = load_a_frag_f32(A, Kpad, rowA, k0, lane);
    v16bf b0 = load_b_frag_nk(Wp, Kpad, colB, k0, lane);
    v16bf b1 = load_b_frag_nk(Wp, Kpad, colB + 16, k0, lane);
    acc0 = wmma_bf16(a, b0, acc0);
    acc1 = wmma_bf16(a, b1, acc1);
  }
  int col0 = n0 + wn + (lane & 15);
  int rowb = m0 + wm + ((lane >> 4) << 3);
#pragma unroll
  for (int e = 0; e < 8; ++e) {
    int row = rowb + e;
    if (col0 < G4)      C[(long)row * G4 + col0]      = acc0[e] + bias[col0];
    if (col0 + 16 < G4) C[(long)row * G4 + col0 + 16] = acc1[e] + bias[col0 + 16];
  }
}

// ---------------- persistent LSTM recurrence ----------------
// Each WG owns 16 batch rows; h (bf16, K-padded 128) and c (f32) live in LDS
// across all 64 timesteps. Per step: G = h * Whh^T via WMMA (25 N-tiles of
// 16x16, K=128 from LDS A-frags + global bf16 B-frags), then fused gates.
__global__ void k_lstm_seq(const float* __restrict__ xp,    // [B,S,400] (incl. bias)
                           const __bf16* __restrict__ whh,  // [400][128] bf16
                           float* __restrict__ hout) {      // [B,S,100]
  __shared__ __bf16 sH[16][128];   // 4 KB
  __shared__ float  sC[16][Hh];    // 6.4 KB
  __shared__ float  sG[16][G4];    // 25.6 KB
  int b0 = blockIdx.x * 16;
  int tid = threadIdx.x, wave = tid >> 5, lane = tid & 31;
  for (int idx = tid; idx < 16 * 128; idx += 256) sH[idx >> 7][idx & 127] = (__bf16)0.f;
  for (int idx = tid; idx < 16 * Hh; idx += 256) sC[idx / Hh][idx % Hh] = 0.f;
  __syncthreads();
  for (int t = 0; t < Sq; ++t) {
    for (int nt = wave; nt < 25; nt += 8) {          // 25 N-tiles of 16
      f32x8 acc = zero8();
      const __bf16* pb = whh + ((long)(nt * 16 + (lane & 15)) << 7) + ((lane >> 4) << 4);
#pragma unroll
      for (int kk = 0; kk < 128; kk += 32) {
        v16bf a = load_a_frag(&sH[0][0], 128, 0, kk, lane);
        v16bf b = *(const v16bf*)(pb + kk);
        acc = wmma_bf16(a, b, acc);
      }
      int col = nt * 16 + (lane & 15);
      int rb = (lane >> 4) << 3;
#pragma unroll
      for (int e = 0; e < 8; ++e) sG[rb + e][col] = acc[e];
    }
    __syncthreads();
    // fused gates, 4 lanes at a time (float4 / bf16x4)
    for (int idx = tid; idx < 16 * 25; idx += 256) {
      int r = idx / 25, jq = (idx % 25) * 4;
      long base = ((long)(b0 + r) * Sq + t) * G4;
      float4 xi = *(const float4*)&xp[base + jq];
      float4 xf = *(const float4*)&xp[base + Hh + jq];
      float4 xg = *(const float4*)&xp[base + 2 * Hh + jq];
      float4 xo = *(const float4*)&xp[base + 3 * Hh + jq];
      float4 gi = *(const float4*)&sG[r][jq];
      float4 gf = *(const float4*)&sG[r][Hh + jq];
      float4 gg = *(const float4*)&sG[r][2 * Hh + jq];
      float4 go = *(const float4*)&sG[r][3 * Hh + jq];
      float4 cc = *(const float4*)&sC[r][jq];
      float4 hh;
      float* pc = (float*)&cc; float* ph = (float*)&hh;
      float* ai = (float*)&gi; float* af = (float*)&gf;
      float* ag = (float*)&gg; float* ao = (float*)&go;
      float* bi = (float*)&xi; float* bfp = (float*)&xf;
      float* bg = (float*)&xg; float* bo = (float*)&xo;
#pragma unroll
      for (int e = 0; e < 4; ++e) {
        float c = sigf(af[e] + bfp[e]) * pc[e] + sigf(ai[e] + bi[e]) * tanhf(ag[e] + bg[e]);
        ph[e] = sigf(ao[e] + bo[e]) * tanhf(c);
        pc[e] = c;
      }
      *(float4*)&sC[r][jq] = cc;
      bf16x4 h4;
#pragma unroll
      for (int e = 0; e < 4; ++e) h4[e] = (__bf16)ph[e];
      *(bf16x4*)&sH[r][jq] = h4;
      *(float4*)&hout[((long)(b0 + r) * Sq + t) * Hh + jq] = hh;
    }
    __syncthreads();
  }
}

// ---------------- elementwise multi-perspective cosine ----------------
// out[b,s,off+k] = cos(w_k*a[b,s], w_k*bside). bside row = s, or lensB[b]-1.
__global__ void k_mp_cos(const float* __restrict__ A, const float* __restrict__ Bf,
                         const int* __restrict__ lensB, const float* __restrict__ W,
                         float* __restrict__ rep, int off) {
  __shared__ float sa[Hh], sb[Hh];
  int bs = blockIdx.x, b = bs >> 6, s = bs & 63;
  int tid = threadIdx.x;
  int brow = lensB ? (lensB[b] - 1) : s;
  for (int h = tid; h < Hh; h += 64) {
    sa[h] = A[(long)bs * Hh + h];
    sb[h] = Bf[((long)b * Sq + brow) * Hh + h];
  }
  __syncthreads();
  if (tid < Kp) {
    const float* w = &W[tid * Hh];
    float num = 0.f, na = 0.f, nb = 0.f;
    for (int h = 0; h < Hh; ++h) {
      float w2 = w[h] * w[h], a = sa[h], bb = sb[h];
      num += a * bb * w2; na += a * a * w2; nb += bb * bb * w2;
    }
    rep[(long)bs * F8K + off + tid] = num / fmaxf(sqrtf(na) * sqrtf(nb), 1e-8f);
  }
}

// ---------------- pairwise mp-cosine + max over j (WMMA) ----------------
// One WG per (b,k). num[i][j] = <a_i*|w_k|, b_j*|w_k|> as 64x64x128 GEMM.
__global__ void k_mp_pool(const float* __restrict__ Af, const float* __restrict__ Bf,
                          const float* __restrict__ W, float* __restrict__ rep, int off) {
  __shared__ __bf16 sA[64][128];
  __shared__ __bf16 sB[64][128];
  __shared__ float  sNum[64][64];
  __shared__ float  sNa[64], sNb[64], sW[128];
  int b = blockIdx.x, k = blockIdx.y;
  int tid = threadIdx.x, wave = tid >> 5, lane = tid & 31;
  for (int h = tid; h < 128; h += 256) sW[h] = (h < Hh) ? fabsf(W[k * Hh + h]) : 0.f;
  __syncthreads();
  for (int idx = tid; idx < 64 * 128; idx += 256) {
    int r = idx >> 7, h = idx & 127;
    float s = sW[h];
    float av = (h < Hh) ? Af[((long)b * Sq + r) * Hh + h] * s : 0.f;
    float bv = (h < Hh) ? Bf[((long)b * Sq + r) * Hh + h] * s : 0.f;
    sA[r][h] = (__bf16)av;
    sB[r][h] = (__bf16)bv;
  }
  __syncthreads();
  if (tid < 128) {
    int r = tid & 63;
    const __bf16* src = (tid < 64) ? &sA[r][0] : &sB[r][0];
    float s = 0.f;
    for (int h = 0; h < Hh; ++h) { float v = (float)src[h]; s += v * v; }
    float n = sqrtf(s);
    if (tid < 64) sNa[r] = n; else sNb[r] = n;
  }
  for (int tileid = wave; tileid < 16; tileid += 8) {
    int mt = (tileid & 3) * 16, nt = (tileid >> 2) * 16;
    f32x8 acc = zero8();
#pragma unroll
    for (int kk = 0; kk < 128; kk += 32)
      acc = wmma_bf16(load_a_frag(&sA[0][0], 128, mt, kk, lane),
                      load_b_frag_nk(&sB[0][0], 128, nt, kk, lane), acc);
    int col = nt + (lane & 15), rb = mt + ((lane >> 4) << 3);
#pragma unroll
    for (int e = 0; e < 8; ++e) sNum[rb + e][col] = acc[e];
  }
  __syncthreads();
  if (tid < 64) {
    float na = sNa[tid], m = -1e30f;
    for (int j = 0; j < 64; ++j)
      m = fmaxf(m, sNum[tid][j] / fmaxf(na * sNb[j], 1e-8f));
    rep[((long)b * Sq + tid) * F8K + off + k] = m;
  }
}

// ---------------- cosine attention + attn-weighted mean + argmax gather ----------------
__global__ void k_attn(const float* __restrict__ Q1, const float* __restrict__ Q2,
                       const int* __restrict__ l1, const int* __restrict__ l2,
                       float* __restrict__ sim, float* __restrict__ gat) {
  __shared__ __bf16 sA[64][128];    // q1 [i][h]
  __shared__ __bf16 sB[64][128];    // q2 [j][h]
  __shared__ __bf16 sBT[112][64];   // q2 [h][j]
  __shared__ __bf16 sAt[64][64];    // attn
  __shared__ float  sNa[64], sNb[64], sSum[64];
  __shared__ int    sArg[64];
  int b = blockIdx.x;
  int tid = threadIdx.x, wave = tid >> 5, lane = tid & 31;
  int L1 = l1[b], L2 = l2[b];
  for (int idx = tid; idx < 64 * 128; idx += 256) {
    int r = idx >> 7, h = idx & 127;
    float a = (h < Hh) ? Q1[((long)b * Sq + r) * Hh + h] : 0.f;
    float v = (h < Hh) ? Q2[((long)b * Sq + r) * Hh + h] : 0.f;
    sA[r][h] = (__bf16)a;
    sB[r][h] = (__bf16)v;
    if (h < 112) sBT[h][r] = (__bf16)v;   // v==0 for h>=100
  }
  if (tid < 128) {
    int r = tid & 63;
    const float* src = (tid < 64) ? Q1 : Q2;
    float s = 0.f;
    for (int h = 0; h < Hh; ++h) { float v = src[((long)b * Sq + r) * Hh + h]; s += v * v; }
    float n = sqrtf(s);
    if (tid < 64) sNa[r] = n; else sNb[r] = n;
  }
  __syncthreads();
  // attn = cos(q1_i, q2_j) * mask
  for (int tileid = wave; tileid < 16; tileid += 8) {
    int mt = (tileid & 3) * 16, nt = (tileid >> 2) * 16;
    f32x8 acc = zero8();
#pragma unroll
    for (int kk = 0; kk < 128; kk += 32)
      acc = wmma_bf16(load_a_frag(&sA[0][0], 128, mt, kk, lane),
                      load_b_frag_nk(&sB[0][0], 128, nt, kk, lane), acc);
    int col = nt + (lane & 15), rb = mt + ((lane >> 4) << 3);
#pragma unroll
    for (int e = 0; e < 8; ++e) {
      int i = rb + e, j = col;
      float cos = acc[e] / fmaxf(sNa[i] * sNb[j], 1e-8f);
      float m = ((i < L1) && (j < L2)) ? 1.f : 0.f;
      sAt[i][j] = (__bf16)(cos * m);
    }
  }
  __syncthreads();
  if (tid < 64) {
    float s = 0.f, best = -1e30f; int bi = 0;
    for (int j = 0; j < 64; ++j) {
      float v = (float)sAt[tid][j];
      s += v;
      if (v > best) { best = v; bi = j; }
    }
    sSum[tid] = s; sArg[tid] = bi;
  }
  __syncthreads();
  // sim = attn * Q2 / (rowsum+eps): 64x112x64 GEMM, A=attn, B from sBT[h][j]
  for (int tileid = wave; tileid < 28; tileid += 8) {
    int mt = (tileid & 3) * 16, nt = (tileid >> 2) * 16;
    f32x8 acc = zero8();
#pragma unroll
    for (int kk = 0; kk < 64; kk += 32)
      acc = wmma_bf16(load_a_frag(&sAt[0][0], 64, mt, kk, lane),
                      load_b_frag_nk(&sBT[0][0], 64, nt, kk, lane), acc);
    int h = nt + (lane & 15), rb = mt + ((lane >> 4) << 3);
    if (h < Hh) {
#pragma unroll
      for (int e = 0; e < 8; ++e) {
        int i = rb + e;
        sim[((long)b * Sq + i) * Hh + h] = acc[e] / (sSum[i] + 1e-8f);
      }
    }
  }
  for (int idx = tid; idx < 64 * Hh; idx += 256) {
    int i = idx / Hh, h = idx % Hh;
    gat[((long)b * Sq + i) * Hh + h] = Q2[((long)b * Sq + sArg[i]) * Hh + h];
  }
}

// ---------------- aggregation + MLP head ----------------
__global__ void k_final(const float* __restrict__ f12, const float* __restrict__ b12,
                        const float* __restrict__ f21, const float* __restrict__ b21,
                        const float* __restrict__ aggW, const float* __restrict__ aggb,
                        const float* __restrict__ predW, const float* __restrict__ predb,
                        float* __restrict__ out) {
  __shared__ float agg[4 * Hh];
  __shared__ float hdn[Hh];
  __shared__ float red[128];
  int b = blockIdx.x, tid = threadIdx.x;
  for (int j = tid; j < Hh; j += 128) {
    float m0 = -1e30f, m1 = -1e30f, m2 = -1e30f, m3 = -1e30f;
    for (int t = 0; t < Sq; ++t) {
      long o = ((long)b * Sq + t) * Hh + j;
      m0 = fmaxf(m0, f12[o]); m1 = fmaxf(m1, b12[o]);
      m2 = fmaxf(m2, f21[o]); m3 = fmaxf(m3, b21[o]);
    }
    agg[j] = m0; agg[Hh + j] = m1; agg[2 * Hh + j] = m2; agg[3 * Hh + j] = m3;
  }
  __syncthreads();
  for (int j = tid; j < Hh; j += 128) {
    float s = aggb[j];
    for (int k = 0; k < 4 * Hh; ++k) s += aggW[j * 4 * Hh + k] * agg[k];
    hdn[j] = tanhf(s);
  }
  __syncthreads();
  float s = 0.f;
  for (int j = tid; j < Hh; j += 128) s += hdn[j] * predW[j];
  red[tid] = s;
  __syncthreads();
  for (int st = 64; st > 0; st >>= 1) {
    if (tid < st) red[tid] += red[tid + st];
    __syncthreads();
  }
  if (tid == 0) out[b] = 1.f / (1.f + expf(-(red[0] + predb[0])));
}

// ---------------- host orchestration ----------------
extern "C" void kernel_launch(void* const* d_in, const int* in_sizes, int n_in,
                              void* d_out, int out_size, void* d_ws, size_t ws_size,
                              hipStream_t stream) {
  const float* emb    = (const float*)d_in[0];
  const float* cfWih  = (const float*)d_in[1];
  const float* cfWhh  = (const float*)d_in[2];
  const float* cfb    = (const float*)d_in[3];
  const float* cbWih  = (const float*)d_in[4];
  const float* cbWhh  = (const float*)d_in[5];
  const float* cbb    = (const float*)d_in[6];
  const float* dfWih  = (const float*)d_in[7];
  const float* dfWhh  = (const float*)d_in[8];
  const float* dfb    = (const float*)d_in[9];
  const float* dbWih  = (const float*)d_in[10];
  const float* dbWhh  = (const float*)d_in[11];
  const float* dbb    = (const float*)d_in[12];
  const float* Wfull  = (const float*)d_in[13];
  const float* Wpool  = (const float*)d_in[14];
  const float* Watt   = (const float*)d_in[15];
  const float* Wmaxa  = (const float*)d_in[16];
  const float* aggW   = (const float*)d_in[17];
  const float* aggb   = (const float*)d_in[18];
  const float* predW  = (const float*)d_in[19];
  const float* predb  = (const float*)d_in[20];
  const int*   q1     = (const int*)d_in[21];
  const int*   q2     = (const int*)d_in[22];
  const int*   l1     = (const int*)d_in[23];
  const int*   l2     = (const int*)d_in[24];
  float* out = (float*)d_out;

  // workspace carve (bump allocator, 256B aligned)
  char* w = (char*)d_ws;
  auto take = [&](size_t bytes) -> char* {
    char* p = w;
    w += (bytes + 255) & ~(size_t)255;
    return p;
  };
  const size_t EB = (size_t)Mrows * EP * 4;
  const size_t XB = (size_t)Mrows * G4 * 4;
  const size_t HB = (size_t)Mrows * Hh * 4;
  const size_t RB = (size_t)Mrows * F8K * 4;
  float* e1    = (float*)take(EB);
  float* e2    = (float*)take(EB);
  float* erev  = (float*)take(EB);
  float* xp    = (float*)take(XB);
  float* q1f   = (float*)take(HB);
  float* q1b   = (float*)take(HB);
  float* q2f   = (float*)take(HB);
  float* q2b   = (float*)take(HB);
  float* tmp   = (float*)take(HB);
  float* simf  = (float*)take(HB);
  float* simb  = (float*)take(HB);
  float* gatf  = (float*)take(HB);
  float* gatb  = (float*)take(HB);
  float* rep12 = (float*)take(RB);
  float* rep21 = (float*)take(RB);
  float* reprev= (float*)take(RB);
  float* f12   = (float*)take(HB);
  float* b12   = (float*)take(HB);
  float* f21   = (float*)take(HB);
  float* b21   = (float*)take(HB);
  __bf16* wihp = (__bf16*)take((size_t)NPAD * EP * 2);   // max Kpad = 320
  __bf16* whhp = (__bf16*)take((size_t)G4 * 128 * 2);

  dim3 blkBS(Mrows), thr128(128), thr256(256), thr64(64);
  dim3 gGemm(Mrows / 64, 7);   // N = 400 -> 7 tiles of 64 (padded weights)

  auto run_lstm = [&](const float* x, int Kact, int Kpad, const float* Wih,
                      const float* bias, const float* Whh, float* hout) {
    k_pack_w<<<dim3(224), thr256, 0, stream>>>(Wih, wihp, G4, Kact, Kpad);
    k_pack_whh<<<dim3(64), thr256, 0, stream>>>(Whh, whhp);
    k_gemm_bias<<<gGemm, thr256, 0, stream>>>(x, wihp, bias, xp, Kpad);
    k_lstm_seq<<<dim3(Bq / 16), thr256, 0, stream>>>(xp, whhp, hout);
  };

  // ---- embeddings (padded to EP) ----
  k_embed<<<blkBS, thr128, 0, stream>>>(q1, emb, e1);
  k_embed<<<blkBS, thr128, 0, stream>>>(q2, emb, e2);

  // ---- contextual BiLSTM (computed once, reused for both directions) ----
  run_lstm(e1, Eemb, EP, cfWih, cfb, cfWhh, q1f);
  k_mask<<<blkBS, thr128, 0, stream>>>(q1f, l1, Hh);
  k_reverse<<<blkBS, thr128, 0, stream>>>(e1, erev, l1, EP);
  run_lstm(erev, Eemb, EP, cbWih, cbb, cbWhh, tmp);
  k_reverse<<<blkBS, thr128, 0, stream>>>(tmp, q1b, l1, Hh);

  run_lstm(e2, Eemb, EP, cfWih, cfb, cfWhh, q2f);
  k_mask<<<blkBS, thr128, 0, stream>>>(q2f, l2, Hh);
  k_reverse<<<blkBS, thr128, 0, stream>>>(e2, erev, l2, EP);
  run_lstm(erev, Eemb, EP, cbWih, cbb, cbWhh, tmp);
  k_reverse<<<blkBS, thr128, 0, stream>>>(tmp, q2b, l2, Hh);

  // ---- matching rep12 (q1 vs q2) ----
  k_mp_cos<<<blkBS, thr64, 0, stream>>>(q1f, q2f, l2, Wfull, rep12, 0);
  k_mp_cos<<<blkBS, thr64, 0, stream>>>(q1b, q2b, l2, Wfull, rep12, 20);
  k_mp_pool<<<dim3(Bq, Kp), thr256, 0, stream>>>(q1f, q2f, Wpool, rep12, 40);
  k_mp_pool<<<dim3(Bq, Kp), thr256, 0, stream>>>(q1b, q2b, Wpool, rep12, 60);
  k_attn<<<dim3(Bq), thr256, 0, stream>>>(q1f, q2f, l1, l2, simf, gatf);
  k_attn<<<dim3(Bq), thr256, 0, stream>>>(q1b, q2b, l1, l2, simb, gatb);
  k_mp_cos<<<blkBS, thr64, 0, stream>>>(q1f, simf, nullptr, Watt, rep12, 80);
  k_mp_cos<<<blkBS, thr64, 0, stream>>>(q1b, simb, nullptr, Watt, rep12, 100);
  k_mp_cos<<<blkBS, thr64, 0, stream>>>(q1f, gatf, nullptr, Wmaxa, rep12, 120);
  k_mp_cos<<<blkBS, thr64, 0, stream>>>(q1b, gatb, nullptr, Wmaxa, rep12, 140);

  // ---- matching rep21 (q2 vs q1) ----
  k_mp_cos<<<blkBS, thr64, 0, stream>>>(q2f, q1f, l1, Wfull, rep21, 0);
  k_mp_cos<<<blkBS, thr64, 0, stream>>>(q2b, q1b, l1, Wfull, rep21, 20);
  k_mp_pool<<<dim3(Bq, Kp), thr256, 0, stream>>>(q2f, q1f, Wpool, rep21, 40);
  k_mp_pool<<<dim3(Bq, Kp), thr256, 0, stream>>>(q2b, q1b, Wpool, rep21, 60);
  k_attn<<<dim3(Bq), thr256, 0, stream>>>(q2f, q1f, l2, l1, simf, gatf);
  k_attn<<<dim3(Bq), thr256, 0, stream>>>(q2b, q1b, l2, l1, simb, gatb);
  k_mp_cos<<<blkBS, thr64, 0, stream>>>(q2f, simf, nullptr, Watt, rep21, 80);
  k_mp_cos<<<blkBS, thr64, 0, stream>>>(q2b, simb, nullptr, Watt, rep21, 100);
  k_mp_cos<<<blkBS, thr64, 0, stream>>>(q2f, gatf, nullptr, Wmaxa, rep21, 120);
  k_mp_cos<<<blkBS, thr64, 0, stream>>>(q2b, gatb, nullptr, Wmaxa, rep21, 140);

  // ---- matching_pools BiLSTM over rep12 / rep21 ----
  run_lstm(rep12, F8K, F8K, dfWih, dfb, dfWhh, f12);
  k_mask<<<blkBS, thr128, 0, stream>>>(f12, l1, Hh);
  k_reverse<<<blkBS, thr128, 0, stream>>>(rep12, reprev, l1, F8K);
  run_lstm(reprev, F8K, F8K, dbWih, dbb, dbWhh, tmp);
  k_reverse<<<blkBS, thr128, 0, stream>>>(tmp, b12, l1, Hh);

  run_lstm(rep21, F8K, F8K, dfWih, dfb, dfWhh, f21);
  k_mask<<<blkBS, thr128, 0, stream>>>(f21, l2, Hh);
  k_reverse<<<blkBS, thr128, 0, stream>>>(rep21, reprev, l2, F8K);
  run_lstm(reprev, F8K, F8K, dbWih, dbb, dbWhh, tmp);
  k_reverse<<<blkBS, thr128, 0, stream>>>(tmp, b21, l2, Hh);

  // ---- head ----
  k_final<<<dim3(Bq), thr128, 0, stream>>>(f12, b12, f21, b21, aggW, aggb, predW, predb, out);
  (void)in_sizes; (void)n_in; (void)out_size; (void)ws_size;
}
